// SelfLinAttention2_10505490006112
// MI455X (gfx1250) — matrix operand order repaired and checked by reading, not compiled, not run on hardware
//
#include <hip/hip_runtime.h>

typedef __attribute__((ext_vector_type(16))) __bf16    v16bf;
typedef __attribute__((ext_vector_type(2)))  __bf16    bf2;
typedef __attribute__((ext_vector_type(8)))  float     v8f;
typedef __attribute__((ext_vector_type(8)))  unsigned  v8u;

constexpr int Bb = 2, Ss = 2048, Hh = 16, Dd = 64;
constexpr int P    = 16;        // sequence partitions -> B*H*P = 512 workgroups
constexpr int ROWS = Ss / P;    // 128 rows per partition
constexpr int CH   = 16;        // chunk rows (one WMMA M-tile)
constexpr int NCH  = ROWS / CH; // 8 chunks per partition
constexpr int NA   = 80;        // augmented N width: 64 V cols + [ones|0...] strip
constexpr int NW   = 5;         // waves per block (one per 16-wide N strip)

// padded LDS leading dims (odd number of dwords -> conflict-free strided reads)
constexpr int LD64 = 66;        // for 64-wide bf16 rows (stride 33 dwords)
constexpr int LD16 = 18;        // for 16-wide bf16 rows (stride 9 dwords)

__device__ __forceinline__ float elu1(float x) { return x > 0.f ? x + 1.f : __expf(x); }

__device__ __forceinline__ unsigned packbf(float a, float b) {
  bf2 t; t[0] = (__bf16)a; t[1] = (__bf16)b;
  return __builtin_bit_cast(unsigned, t);
}

// ---------------- WMMA bf16 operand gathers (wave32 layouts) ----------------
// A (16xK): lane -> row m, half = lane>>4; pair r (elems 2r,2r+1):
//   k_even = 16*(r>=4) + 8*half + 2*(r&3).  src row-major along k, ld even.
template<int KVALID>
__device__ __forceinline__ v16bf ldA(const __bf16* src, int ld, int m, int half, int kbase) {
  v8u u;
#pragma unroll
  for (int r = 0; r < 8; ++r) {
    int k = ((r & 4) << 2) + 8 * half + ((r & 3) << 1);
    u[r] = (k < KVALID) ? *(const unsigned*)(src + m * ld + kbase + k) : 0u;
  }
  return __builtin_bit_cast(v16bf, u);
}

// B (Kx16): lane -> col n, khalf = 16*(lane>>4); pair r: k_even = 16*(r>=4) + khalf/2 + 2*(r&3).
// src stored k-contiguous at row n (i.e. transposed storage), ld even.
template<int KVALID>
__device__ __forceinline__ v16bf ldB(const __bf16* src, int ld, int n, int khalf, int kbase) {
  v8u u;
#pragma unroll
  for (int r = 0; r < 8; ++r) {
    int k = ((r & 4) << 2) + (khalf >> 1) + ((r & 3) << 1);
    u[r] = (k < KVALID) ? *(const unsigned*)(src + n * ld + kbase + k) : 0u;
  }
  return __builtin_bit_cast(v16bf, u);
}

#define WMMA_BF16(A, B, C) \
  __builtin_amdgcn_wmma_f32_16x16x32_bf16(false, (A), false, (B), (short)0, (C), false, false)

// ===== Kernel 1: per-partition augmented state KV_aug = K^T [V | 1 | 0] =====
__global__ void __launch_bounds__(32 * NW)
lin_partial_kernel(const float* __restrict__ qk, const float* __restrict__ vin,
                   float* __restrict__ wsKV) {
  __shared__ __align__(16) __bf16 sKT[Dd * LD16];   // [d][j]
  __shared__ __align__(16) __bf16 sVT[NA * LD16];   // [t][j], rows 64..79 constant

  const int bhp = blockIdx.x;
  const int p = bhp % P, bh = bhp / P;
  const int b = bh / Hh, h = bh % Hh;
  const int tid = threadIdx.x;
  const int w = tid >> 5, lane = tid & 31;
  const int n = lane & 15, half = lane >> 4;
  const int row = tid >> 3, d0 = (tid & 7) * 8;   // loader mapping (tid < 128)

  // augmentation strip of V: row 64 = ones, rows 65..79 = zeros (constant)
  if (w == 4) {
    int rr = lane >> 1, j0 = (lane & 1) * 8;
#pragma unroll
    for (int e = 0; e < 8; ++e)
      sVT[(64 + rr) * LD16 + j0 + e] = (rr == 0) ? (__bf16)1.0f : (__bf16)0.0f;
  }

  v8f kv[4] = {};

  for (int c = 0; c < NCH; ++c) {
    const int s0 = p * ROWS + c * CH;
    __syncthreads();
    if (tid < 128) {
      int ik = (((b * Ss + (s0 + row)) * 2 + 1) * Hh + h) * Dd + d0;
      int iv = ((b * Ss + (s0 + row)) * Hh + h) * Dd + d0;
      float4 ka = *(const float4*)(qk + ik);
      float4 kb = *(const float4*)(qk + ik + 4);
      float4 va = *(const float4*)(vin + iv);
      float4 vb = *(const float4*)(vin + iv + 4);
      float ke[8] = {elu1(ka.x), elu1(ka.y), elu1(ka.z), elu1(ka.w),
                     elu1(kb.x), elu1(kb.y), elu1(kb.z), elu1(kb.w)};
      float ve[8] = {va.x, va.y, va.z, va.w, vb.x, vb.y, vb.z, vb.w};
#pragma unroll
      for (int e = 0; e < 8; ++e) {
        sKT[(d0 + e) * LD16 + row] = (__bf16)ke[e];
        sVT[(d0 + e) * LD16 + row] = (__bf16)ve[e];
      }
      if (c + 1 < NCH) {
        __builtin_prefetch(qk + ik + CH * 2 * Hh * Dd, 0, 1);
        __builtin_prefetch(vin + iv + CH * Hh * Dd, 0, 1);
      }
    }
    __syncthreads();

    // KV_aug += K^T V_aug  (each wave owns a 16-wide strip; 4 M-tiles over d)
    v16bf bV = ldB<CH>(sVT, LD16, 16 * w + n, 16 * half, 0);
#pragma unroll
    for (int mt = 0; mt < 4; ++mt) {
      v16bf aKt = ldA<CH>(sKT, LD16, 16 * mt + (lane & 15), half, 0);
      kv[mt] = WMMA_BF16(aKt, bV, kv[mt]);
    }
  }

  float* dstKV = wsKV + (size_t)bhp * (Dd * NA);   // [d][t_aug] fp32
#pragma unroll
  for (int mt = 0; mt < 4; ++mt)
#pragma unroll
    for (int r = 0; r < 8; ++r)
      dstKV[(16 * mt + r + 8 * half) * NA + 16 * w + n] = kv[mt][r];
}

// ============ Kernel 2: exclusive prefix of state + chunked scan ============
__global__ void __launch_bounds__(32 * NW)
lin_main_kernel(const float* __restrict__ qk, const float* __restrict__ vin,
                float* __restrict__ outp, const float* __restrict__ wsKV) {
  __shared__ __align__(16) __bf16 sQ [CH * LD64];   // [i][d]
  __shared__ __align__(16) __bf16 sK [CH * LD64];   // [j][d]   (B for Q K^T)
  __shared__ __align__(16) __bf16 sKT[Dd * LD16];   // [d][j]   (A for K^T V)
  __shared__ __align__(16) __bf16 sVT[NA * LD16];   // [t][j]   (B operand, augmented)
  __shared__ __align__(16) __bf16 sKVT[NA * LD64];  // [t][d]   (B for Q*KV, augmented)
  __shared__ __align__(16) __bf16 sS [CH * LD16];   // [i][j]   masked scores
  __shared__ float  sKVf[Dd * NA];                  // fp32 init state [d][t_aug]
  __shared__ float  sInvN[CH];

  const int bhp = blockIdx.x;
  const int p = bhp % P, bh = bhp / P;
  const int b = bh / Hh, h = bh % Hh;
  const int tid = threadIdx.x;
  const int w = tid >> 5, lane = tid & 31;
  const int n = lane & 15, half = lane >> 4;
  const int row = tid >> 3, d0 = (tid & 7) * 8;    // loader mapping (tid < 128)

  // augmentation strip of V (constant across chunks)
  if (w == 4) {
    int rr = lane >> 1, j0 = (lane & 1) * 8;
#pragma unroll
    for (int e = 0; e < 8; ++e)
      sVT[(64 + rr) * LD16 + j0 + e] = (rr == 0) ? (__bf16)1.0f : (__bf16)0.0f;
  }

  // ---- exclusive prefix over earlier partitions -> initial augmented state ----
  for (int j = 0; j < 32; ++j) {                    // 5120 = 160*32 elements
    int idx = tid + 160 * j;
    float acc = 0.f;
    for (int pp = 0; pp < p; ++pp) acc += wsKV[(size_t)(bh * P + pp) * (Dd * NA) + idx];
    sKVf[idx] = acc;
  }
  __syncthreads();

  v8f kv[4];
#pragma unroll
  for (int mt = 0; mt < 4; ++mt)
#pragma unroll
    for (int r = 0; r < 8; ++r)
      kv[mt][r] = sKVf[(16 * mt + r + 8 * half) * NA + 16 * w + n];
  for (int j = 0; j < 32; ++j) {                    // sKVT[t][d] = KV[d][t]
    int idx = tid + 160 * j;
    int t = idx >> 6, d = idx & 63;
    sKVT[t * LD64 + d] = (__bf16)sKVf[d * NA + t];
  }
  // visibility of sKVT/sVT covered by the barrier at the top of the loop

  for (int c = 0; c < NCH; ++c) {
    const int s0 = p * ROWS + c * CH;
    __syncthreads();                               // A: protect chunk bufs + sKVT
    if (tid < 128) {
      int iq = (((b * Ss + (s0 + row)) * 2 + 0) * Hh + h) * Dd + d0;
      int ik = iq + Hh * Dd;
      int iv = ((b * Ss + (s0 + row)) * Hh + h) * Dd + d0;
      float4 qa = *(const float4*)(qk + iq);
      float4 qb = *(const float4*)(qk + iq + 4);
      float4 ka = *(const float4*)(qk + ik);
      float4 kb = *(const float4*)(qk + ik + 4);
      float4 va = *(const float4*)(vin + iv);
      float4 vb = *(const float4*)(vin + iv + 4);
      float qe[8] = {elu1(qa.x), elu1(qa.y), elu1(qa.z), elu1(qa.w),
                     elu1(qb.x), elu1(qb.y), elu1(qb.z), elu1(qb.w)};
      float ke[8] = {elu1(ka.x), elu1(ka.y), elu1(ka.z), elu1(ka.w),
                     elu1(kb.x), elu1(kb.y), elu1(kb.z), elu1(kb.w)};
      float ve[8] = {va.x, va.y, va.z, va.w, vb.x, vb.y, vb.z, vb.w};
#pragma unroll
      for (int i2 = 0; i2 < 4; ++i2) {
        *(unsigned*)&sQ[row * LD64 + d0 + 2 * i2] = packbf(qe[2 * i2], qe[2 * i2 + 1]);
        *(unsigned*)&sK[row * LD64 + d0 + 2 * i2] = packbf(ke[2 * i2], ke[2 * i2 + 1]);
      }
#pragma unroll
      for (int e = 0; e < 8; ++e) {
        sKT[(d0 + e) * LD16 + row] = (__bf16)ke[e];
        sVT[(d0 + e) * LD16 + row] = (__bf16)ve[e];
      }
      if (c + 1 < NCH) {
        __builtin_prefetch(qk + iq + CH * 2 * Hh * Dd, 0, 1);
        __builtin_prefetch(vin + iv + CH * Hh * Dd, 0, 1);
      }
    }
    __syncthreads();                               // B: chunk data + prev state ready

    // Q A-fragments (reused by inter-chunk matmul and wave0 score matmul)
    v16bf aQ0 = ldA<32>(sQ, LD64, lane & 15, half, 0);
    v16bf aQ1 = ldA<32>(sQ, LD64, lane & 15, half, 32);

    // inter-chunk: out_aug = Q * KV_aug_prev  (col 64 -> q . Ksum_prev)
    v8f oacc = {};
    {
      v16bf bKV0 = ldB<32>(sKVT, LD64, 16 * w + n, 16 * half, 0);
      oacc = WMMA_BF16(aQ0, bKV0, oacc);
      v16bf bKV1 = ldB<32>(sKVT, LD64, 16 * w + n, 16 * half, 32);
      oacc = WMMA_BF16(aQ1, bKV1, oacc);
    }

    // state update: KV_aug += K^T V_aug  (fp32 accumulators; col 64 -> Ksum)
    v16bf bV = ldB<CH>(sVT, LD16, 16 * w + n, 16 * half, 0);
#pragma unroll
    for (int mt = 0; mt < 4; ++mt) {
      v16bf aKt = ldA<CH>(sKT, LD16, 16 * mt + (lane & 15), half, 0);
      kv[mt] = WMMA_BF16(aKt, bV, kv[mt]);
    }

    if (w == 0) {
      // intra-chunk scores S = Q K^T with causal mask (j <= i)
      v8f sc = {};
      {
        v16bf bKt0 = ldB<32>(sK, LD64, lane & 15, 16 * half, 0);
        sc = WMMA_BF16(aQ0, bKt0, sc);
        v16bf bKt1 = ldB<32>(sK, LD64, lane & 15, 16 * half, 32);
        sc = WMMA_BF16(aQ1, bKt1, sc);
      }
#pragma unroll
      for (int r = 0; r < 8; ++r) {
        int m = r + 8 * half;
        float sv = (n <= m) ? sc[r] : 0.f;
        sS[m * LD16 + n] = (__bf16)sv;
      }
    }

    __syncthreads();                               // C: sS ready; state reads done

    // intra-chunk: out_aug += (masked S) * V_aug  (col 64 -> +rowsum(S) => n_i)
    {
      v16bf aS = ldA<16>(sS, LD16, lane & 15, half, 0);
      oacc = WMMA_BF16(aS, bV, oacc);
    }

    // wave 4 owns the augmentation strip: col 64 of oacc is the normalizer
    if (w == 4 && n == 0) {
#pragma unroll
      for (int r = 0; r < 8; ++r) sInvN[r + 8 * half] = 1.0f / oacc[r];
    }

    __syncthreads();                               // D: sInvN ready

    if (w < 4) {
#pragma unroll
      for (int r = 0; r < 8; ++r) {
        int m = r + 8 * half;
        float o = oacc[r] * sInvN[m];
        outp[((b * Ss + (s0 + m)) * Hh + h) * Dd + 16 * w + n] = o;
      }
    }
    // refresh bf16 KV^T copy for next chunk's Q*KV (packed pair stores)
#pragma unroll
    for (int mt = 0; mt < 4; ++mt)
#pragma unroll
      for (int rr = 0; rr < 4; ++rr) {
        int d = 16 * mt + 8 * half + 2 * rr;
        *(unsigned*)&sKVT[(16 * w + n) * LD64 + d] = packbf(kv[mt][2 * rr], kv[mt][2 * rr + 1]);
      }
  }
}

extern "C" void kernel_launch(void* const* d_in, const int* in_sizes, int n_in,
                              void* d_out, int out_size, void* d_ws, size_t ws_size,
                              hipStream_t stream) {
  (void)in_sizes; (void)n_in; (void)out_size; (void)ws_size;
  const float* qk  = (const float*)d_in[0];
  const float* vin = (const float*)d_in[1];
  float* outp = (float*)d_out;

  float* wsKV = (float*)d_ws;                      // [B*H*P][64*80] fp32 aug state

  dim3 grid(Bb * Hh * P), blk(32 * NW);
  lin_partial_kernel<<<grid, blk, 0, stream>>>(qk, vin, wsKV);
  lin_main_kernel<<<grid, blk, 0, stream>>>(qk, vin, outp, wsKV);
}
